// LSTMCell_86792699117977
// MI455X (gfx1250) — matrix operand order repaired and checked by reading, not compiled, not run on hardware
//
#include <hip/hip_runtime.h>

// ---- CDNA5 WMMA types -------------------------------------------------------
typedef __bf16 bf16;
typedef __attribute__((ext_vector_type(16))) __bf16        v16bf;
typedef __attribute__((ext_vector_type(8)))  float         v8f;
typedef __attribute__((ext_vector_type(4)))  float         v4f;
typedef __attribute__((ext_vector_type(4)))  unsigned int  v4u;
typedef __attribute__((ext_vector_type(2)))  unsigned int  v2u;

#define BATCH 16384
#define HID   1024

// Block tile: 128 batch rows x 32 hidden cols x all 4 gates, K-step 32 (bf16 WMMA K)
// 512 threads = 16 waves: wave = (wm 0..3, gate 0..3); each wave: 2x2 WMMA tiles (32x32)
#define BM 128
#define BJ 32
#define BK 32
#define LD 40                        // padded LDS row stride (bf16 elems): 80B, 16B aligned
#define A_BYTES   (BM * LD * 2)      // 10240 B
#define B_BYTES   (4 * BJ * LD * 2)  // 10240 B
#define BUF_BYTES (A_BYTES + B_BYTES)
#define NSTEP 64                     // (I + H) / BK = 2048 / 32
#define SMEM_BYTES 65536             // max(2*BUF_BYTES=40960, gate buffer 4*128*32*4=65536)

union FragU { v4u q[2]; v16bf v; };

__device__ __forceinline__ unsigned pk2(float x, float y) {
    // round-to-nearest-even fp32 -> bf16, packed pair
    unsigned ux = __float_as_uint(x), uy = __float_as_uint(y);
    ux += 0x7FFFu + ((ux >> 16) & 1u);
    uy += 0x7FFFu + ((uy >> 16) & 1u);
    return (ux >> 16) | (uy & 0xFFFF0000u);
}

__device__ __forceinline__ float sigmoidf_(float v) { return 1.0f / (1.0f + __expf(-v)); }

__global__ __launch_bounds__(512, 1)
void lstm_cell_wmma(const float* __restrict__ x,   const float* __restrict__ h_prev,
                    const float* __restrict__ c_prev,
                    const float* __restrict__ Wii, const float* __restrict__ Wif,
                    const float* __restrict__ Wig, const float* __restrict__ Wio,
                    const float* __restrict__ bii, const float* __restrict__ bif,
                    const float* __restrict__ big, const float* __restrict__ bio,
                    const float* __restrict__ Whi, const float* __restrict__ Whf,
                    const float* __restrict__ Whg, const float* __restrict__ Who,
                    float* __restrict__ out_h, float* __restrict__ out_c)
{
    extern __shared__ char smem[];
    const int tid  = threadIdx.x;
    const int lane = tid & 31;
    const int wave = tid >> 5;        // 16 waves
    const int gate = wave & 3;        // gate handled by this wave
    const int wm   = wave >> 2;       // 0..3 : which 32-row quarter
    const int m0   = blockIdx.y * BM; // batch tile origin
    const int j0   = blockIdx.x * BJ; // hidden tile origin

    v8f acc[2][2];
    const v8f vzero = {0.f,0.f,0.f,0.f,0.f,0.f,0.f,0.f};
#pragma unroll
    for (int t = 0; t < 2; ++t)
#pragma unroll
        for (int u = 0; u < 2; ++u) acc[t][u] = vzero;

    // ---- loader geometry (512 threads) -------------------------------------
    const int ar = tid >> 3;          // 0..63 : A tile row (and +64 for rep 1)
    const int ac = (tid & 7) << 2;    // 0..28 : k-column (float4 granule)
    const int br = (tid >> 3) & 31;   // 0..31 : weight tile row
    const int g0 = (tid >> 8) & 1;    // rep0 covers gate g0, rep1 covers gate 2+g0

    // Per-thread weight pointers: selected ONCE here, loop bodies are select-free.
    const float* Wi01 = g0 ? Wif : Wii;
    const float* Wi23 = g0 ? Wio : Wig;
    const float* Wh01 = g0 ? Whf : Whi;
    const float* Wh23 = g0 ? Who : Whg;

    // stage a 32-wide K slice: A(128x32) + 4 gate-weight tiles (32x32 each)
    auto load_tile = [&](const float* Ap, const float* W01, const float* W23,
                         int ks, v4f aR[2], v4f bR[2]) {
        const int kk = ks << 5;                   // k offset within phase (0..992)
        aR[0] = *(const v4f*)(Ap  + (m0 + ar)      * HID + kk + ac);
        aR[1] = *(const v4f*)(Ap  + (m0 + ar + 64) * HID + kk + ac);
        bR[0] = *(const v4f*)(W01 + (j0 + br)      * HID + kk + ac);
        bR[1] = *(const v4f*)(W23 + (j0 + br)      * HID + kk + ac);
    };

    auto store_tile = [&](int s, const v4f aR[2], const v4f bR[2]) {
        char* buf = smem + (s & 1) * BUF_BYTES;
        bf16* As = (bf16*)buf;
        bf16* Bs = (bf16*)(buf + A_BYTES);
        { v2u p; p.x = pk2(aR[0].x, aR[0].y); p.y = pk2(aR[0].z, aR[0].w);
          *(v2u*)(As + ar * LD + ac) = p; }
        { v2u p; p.x = pk2(aR[1].x, aR[1].y); p.y = pk2(aR[1].z, aR[1].w);
          *(v2u*)(As + (ar + 64) * LD + ac) = p; }
        { v2u p; p.x = pk2(bR[0].x, bR[0].y); p.y = pk2(bR[0].z, bR[0].w);
          *(v2u*)(Bs + (g0 * BJ + br) * LD + ac) = p; }
        { v2u p; p.x = pk2(bR[1].x, bR[1].y); p.y = pk2(bR[1].z, bR[1].w);
          *(v2u*)(Bs + ((2 + g0) * BJ + br) * LD + ac) = p; }
    };

    // ---- per-lane fragment geometry (ISA 16-bit A 16x32 / B 32x16 layouts) --
    const int amr = lane & 15;          // A: matrix row
    const int akb = (lane >> 4) << 3;   // A: lanes 16-31 start at K=8
    const int bnc = lane & 15;          // B: column (n)
    const int bkb = (lane >> 4) << 3;   // B: lanes 16-31 start at K=8

    auto compute_step = [&](int s) {
        const char* buf = smem + (s & 1) * BUF_BYTES;
        const bf16* As = (const bf16*)buf;
        const bf16* Bs = (const bf16*)(buf + A_BYTES);
        v16bf af[2], bfr[2];
#pragma unroll
        for (int t = 0; t < 2; ++t) {   // A elems: [akb..akb+7] and [akb+16..akb+23]
            const bf16* p = As + (wm * 32 + t * 16 + amr) * LD + akb;
            FragU f; f.q[0] = *(const v4u*)p; f.q[1] = *(const v4u*)(p + 16);
            af[t] = f.v;
        }
#pragma unroll
        for (int u = 0; u < 2; ++u) {   // B elems: [bkb..bkb+7] and [bkb+16..bkb+23]
            const bf16* p = Bs + (gate * BJ + u * 16 + bnc) * LD + bkb;
            FragU f; f.q[0] = *(const v4u*)p; f.q[1] = *(const v4u*)(p + 16);
            bfr[u] = f.v;
        }
#pragma unroll
        for (int t = 0; t < 2; ++t)
#pragma unroll
            for (int u = 0; u < 2; ++u)
                acc[t][u] = __builtin_amdgcn_wmma_f32_16x16x32_bf16(
                    false, af[t], false, bfr[u], (short)0, acc[t][u], false, false);
    };

    // 31 steady-state steps of one phase: load ks+1, store to other buffer,
    // compute ks, one barrier. store(s+1)->buf[(s+1)&1] before compute(s) is
    // safe: that buffer was last read in step s-1, sealed by the prior barrier.
    auto run_phase = [&](const float* Ap, const float* W01, const float* W23, int sbase) {
        for (int ks = 0; ks < 31; ++ks) {
            v4f aN[2], bN[2];
            load_tile(Ap, W01, W23, ks + 1, aN, bN);
            store_tile(sbase + ks + 1, aN, bN);
            // global_prefetch hint, wrap-around offset avoids a loop-splitting guard
            __builtin_prefetch(Ap + (m0 + ar) * HID + (((ks + 2) & 31) << 5) + ac, 0, 1);
            compute_step(sbase + ks);
            __syncthreads();
        }
    };

    // ---- K pipeline: phase 0 (x * Wi^T), phase 1 (h * Wh^T) -----------------
    {
        v4f aR[2], bR[2];
        load_tile(x, Wi01, Wi23, 0, aR, bR);
        store_tile(0, aR, bR);
    }
    __syncthreads();
    run_phase(x, Wi01, Wi23, 0);
    {   // phase boundary: stage h/Wh k=0 while computing s=31
        v4f aR[2], bR[2];
        load_tile(h_prev, Wh01, Wh23, 0, aR, bR);
        store_tile(32, aR, bR);
        compute_step(31);
        __syncthreads();
    }
    run_phase(h_prev, Wh01, Wh23, 32);
    compute_step(63);
    __syncthreads();   // all waves done reading tile buffers before gb overwrite

    // ---- epilogue: gates -> LDS [4][BM][BJ] f32, then fused LSTM pointwise --
    float* gb = (float*)smem;
#pragma unroll
    for (int t = 0; t < 2; ++t)
#pragma unroll
        for (int u = 0; u < 2; ++u)
#pragma unroll
            for (int r = 0; r < 8; ++r) {
                // C/D layout: lanes 0-15 -> M=r, lanes 16-31 -> M=r+8; N = lane&15
                int m = wm * 32 + t * 16 + r + ((lane >> 4) << 3);
                int j = u * 16 + (lane & 15);
                gb[(gate * BM + m) * BJ + j] = acc[t][u][r];
            }
    __syncthreads();

#pragma unroll
    for (int rep = 0; rep < 8; ++rep) {
        int idx = tid + 512 * rep;       // 128*32 = 4096 cells / block
        int m = idx >> 5, j = idx & 31;
        int jj = j0 + j;
        float gi = gb[(0 * BM + m) * BJ + j] + bii[jj];
        float gf = gb[(1 * BM + m) * BJ + j] + bif[jj];
        float gg = gb[(2 * BM + m) * BJ + j] + big[jj];
        float go = gb[(3 * BM + m) * BJ + j] + bio[jj];
        float it = sigmoidf_(gi);
        float ft = sigmoidf_(gf);
        float gt = tanhf(gg);
        float ot = sigmoidf_(go);
        int gidx = (m0 + m) * HID + jj;
        float cn = ft * c_prev[gidx] + it * gt;
        out_c[gidx] = cn;
        out_h[gidx] = ot * tanhf(cn);
    }
}

// ---- host glue --------------------------------------------------------------
extern "C" void kernel_launch(void* const* d_in, const int* in_sizes, int n_in,
                              void* d_out, int out_size, void* d_ws, size_t ws_size,
                              hipStream_t stream) {
    (void)in_sizes; (void)n_in; (void)d_ws; (void)ws_size; (void)out_size;
    const float* x   = (const float*)d_in[0];
    const float* h   = (const float*)d_in[1];
    const float* c   = (const float*)d_in[2];
    const float* Wii = (const float*)d_in[3];
    const float* Wif = (const float*)d_in[4];
    const float* Wig = (const float*)d_in[5];
    const float* Wio = (const float*)d_in[6];
    const float* bii = (const float*)d_in[7];
    const float* bif = (const float*)d_in[8];
    const float* big = (const float*)d_in[9];
    const float* bio = (const float*)d_in[10];
    const float* Whi = (const float*)d_in[11];
    const float* Whf = (const float*)d_in[12];
    const float* Whg = (const float*)d_in[13];
    const float* Who = (const float*)d_in[14];
    float* out_h = (float*)d_out;
    float* out_c = out_h + (size_t)BATCH * HID;

    dim3 grid(HID / BJ, BATCH / BM);   // (32, 128) = 4096 blocks
    lstm_cell_wmma<<<grid, dim3(512), SMEM_BYTES, stream>>>(
        x, h, c, Wii, Wif, Wig, Wio, bii, bif, big, bio,
        Whi, Whf, Whg, Who, out_h, out_c);
}
